// CAUSAL_GAT_20031727469167
// MI455X (gfx1250) — matrix operand order repaired and checked by reading, not compiled, not run on hardware
//
#include <hip/hip_runtime.h>

// ---------------------------------------------------------------------------
// CAUSAL GAT pipeline for gfx1250 (MI455X), wave32.
// B=128, N=50, F_IN=D=64, E=2450, S=49, BN=6400.
// ---------------------------------------------------------------------------
#define Bc    128
#define Nc    50
#define Dc    64
#define BNc   (Bc*Nc)         // 6400
#define NMASK 50              // S+1 submodel masks
#define Sc    49
#define EPSc  1e-5f
#define NEGc  (-1.0e9f)
// pval = betainc(127, 0.5, df2/(df2+f)) > 0.05  <=>  f < F_{0.95}(1,254)
#define FCRIT 3.878324f

typedef __attribute__((ext_vector_type(16))) _Float16 v16h;
typedef __attribute__((ext_vector_type(8)))  float    v8f;
typedef __attribute__((ext_vector_type(4)))  unsigned int v4u;
typedef __attribute__((ext_vector_type(8)))  unsigned int v8u;

__device__ __forceinline__ float wsum(float v){
#pragma unroll
  for (int o = 16; o; o >>= 1) v += __shfl_xor(v, o, 32);
  return v;
}
__device__ __forceinline__ float wmaxr(float v){
#pragma unroll
  for (int o = 16; o; o >>= 1) v = fmaxf(v, __shfl_xor(v, o, 32));
  return v;
}
__device__ __forceinline__ float lrelu(float x){ return x >= 0.f ? x : 0.2f*x; }

// ---------------------------------------------------------------------------
// Tensor Data Mover: DMA `nelem` f32 (contiguous) from global -> LDS.
// Builds a D# (group0: count=1/lds/global/type=2; group1: data_size=4B,
// 1 x nelem tile) in SGPRs and issues TENSOR_LOAD_TO_LDS (VADDR2/3 = NULL).
// Caller must issue from ONE wave and then s_wait_tensorcnt + barrier.
// ---------------------------------------------------------------------------
__device__ __forceinline__ void tdm_load_f32(const float* gsrc, void* lds_dst,
                                             unsigned int nelem /* <= 65535 */){
  unsigned long long ga = (unsigned long long)(uintptr_t)gsrc;
  unsigned int lo = (unsigned int)(uintptr_t)lds_dst;   // LDS byte offset
  v4u g0;
  g0[0] = 1u;                                           // count=1, no gather
  g0[1] = lo;                                           // lds_addr
  g0[2] = (unsigned int)ga;                             // global_addr[31:0]
  g0[3] = ((unsigned int)(ga >> 32) & 0x01FFFFFFu)      // global_addr[56:32]
        | 0x80000000u;                                  // type=2 (image)
  v8u g1;
  g1[0] = 0x20000u;                                     // data_size=2 (4 bytes)
  g1[1] = (nelem & 0xFFFFu) << 16;                      // tensor_dim0[15:0]
  g1[2] = (nelem >> 16) | (1u << 16);                   // dim0 hi | tensor_dim1=1
  g1[3] = (nelem << 16);                                // tile_dim0 = nelem
  g1[4] = 1u;                                           // tile_dim1=1, tile_dim2=0
  g1[5] = nelem;                                        // tensor_dim0_stride lo
  g1[6] = (nelem & 0xFFFFu) << 16;                      // stride0 hi=0 | stride1 lo
  g1[7] = (nelem >> 16);                                // stride1 hi
  asm volatile("tensor_load_to_lds %0, %1" :: "s"(g0), "s"(g1) : "memory");
}

// --------------------------------------------------------------------------
// eI[p] = emb[p] . att_em_i ; eJ[p] = emb[p] . att_em_j
// --------------------------------------------------------------------------
__global__ void k_embdot(const float* __restrict__ emb,
                         const float* __restrict__ aei,
                         const float* __restrict__ aej,
                         float* __restrict__ eI, float* __restrict__ eJ){
  int t = threadIdx.x;
  float si = 0.f, sj = 0.f;
  if (t < Nc){
#pragma unroll
    for (int d = 0; d < Dc; ++d){
      float e = emb[t*Dc + d];
      si += e * aei[d];
      sj += e * aej[d];
    }
  }
  if (t < 64){ eI[t] = si; eJ[t] = sj; }
}

// --------------------------------------------------------------------------
// h = x (6400x64) @ W (64x64), f32 in -> f16 WMMA -> f32 out.
// One wave computes a 16x64 output STRIP: A loaded once, reused across the
// 4 column tiles -> 8 v_wmma_f32_16x16x32_f16 per wave, 4x less x traffic.
// grid 100 blocks x 128 threads (4 waves) = 400 row strips.
// --------------------------------------------------------------------------
__global__ void k_gemm_h(const float* __restrict__ x,
                         const float* __restrict__ W,
                         float* __restrict__ h){
  int wid  = threadIdx.x >> 5;
  int lane = threadIdx.x & 31;
  int tm   = blockIdx.x * 4 + wid;     // 0..399 row strip
  int half = lane >> 4;
  int l16  = lane & 15;

  int row = tm*16 + l16;               // A-matrix: M = lane%16
  const float* xr = x + row*Dc;
  __builtin_prefetch(xr, 0, 3);        // global_prefetch_b8

  // A (16x64 f16) once: two K=32 operand sets
  v16h a0 = {}, a1 = {};
#pragma unroll
  for (int v = 0; v < 8; ++v){
    int kk = ((v < 4) ? 0 : 16) + half*8 + 2*(v & 3);
    a0[2*v]   = (_Float16)xr[kk];
    a0[2*v+1] = (_Float16)xr[kk + 1];
    a1[2*v]   = (_Float16)xr[kk + 32];
    a1[2*v+1] = (_Float16)xr[kk + 33];
  }

#pragma unroll
  for (int tn = 0; tn < 4; ++tn){
    int n = tn*16 + l16;               // B/C/D: N = lane%16
    v16h bb0 = {}, bb1 = {};
#pragma unroll
    for (int v = 0; v < 8; ++v){
      // B 32x16 layout: same K order as A (lanes 0-15 hold K=0..7,16..23; lanes 16-31 hold K=8..15,24..31)
      int kb = ((v < 4) ? 0 : 16) + half*8 + 2*(v & 3);
      bb0[2*v]   = (_Float16)W[(kb     )*Dc + n];
      bb0[2*v+1] = (_Float16)W[(kb +  1)*Dc + n];
      bb1[2*v]   = (_Float16)W[(kb + 32)*Dc + n];
      bb1[2*v+1] = (_Float16)W[(kb + 33)*Dc + n];
    }
    v8f c = {};
    c = __builtin_amdgcn_wmma_f32_16x16x32_f16(false, a0, false, bb0, (short)0, c, false, false);
    c = __builtin_amdgcn_wmma_f32_16x16x32_f16(false, a1, false, bb1, (short)0, c, false, false);
#pragma unroll
    for (int r = 0; r < 8; ++r){
      int orow = tm*16 + r + 8*half;   // C/D layout: VGPR r -> M=r (+8 for hi half)
      h[orow*Dc + tn*16 + l16] = c[r];
    }
  }
}

// --------------------------------------------------------------------------
// sI[i] = h[i].att_i + eI[i%N] ; sJ[i] = h[i].att_j + eJ[i%N]
// --------------------------------------------------------------------------
__global__ void k_sij(const float* __restrict__ h,
                      const float* __restrict__ ai, const float* __restrict__ aj,
                      const float* __restrict__ eI, const float* __restrict__ eJ,
                      float* __restrict__ sI, float* __restrict__ sJ){
  int i = blockIdx.x*256 + threadIdx.x;
  if (i >= BNc) return;
  int p = i % Nc;
  float si = eI[p], sj = eJ[p];
  const float* hr = h + i*Dc;
#pragma unroll
  for (int d = 0; d < Dc; ++d){ float v = hr[d]; si += v*ai[d]; sj += v*aj[d]; }
  sI[i] = si; sJ[i] = sj;
}

// --------------------------------------------------------------------------
// 50 inference passes, fully fused: masked softmax attention + aggregation +
// BN1(inference) + relu + *emb + BN2(inference) + relu + out head.
// One block per (mask mi, batch b); h_b DMA'd into LDS by the TDM (wave 0
// issues the descriptor, waits TENSORcnt, barrier releases the block);
// 8 waves cover p=0..49. Mask mi>0 removes src q=(mi-1<p ? mi-1 : mi).
// --------------------------------------------------------------------------
__global__ void k_preds(const float* __restrict__ h,  const float* __restrict__ sI,
                        const float* __restrict__ sJ, const float* __restrict__ gbias,
                        const float* __restrict__ g1, const float* __restrict__ b1,
                        const float* __restrict__ rm1,const float* __restrict__ rv1,
                        const float* __restrict__ g2, const float* __restrict__ b2,
                        const float* __restrict__ rm2,const float* __restrict__ rv2,
                        const float* __restrict__ emb,const float* __restrict__ outw,
                        const float* __restrict__ outb, float* __restrict__ preds){
  __shared__ float sh[Nc*Dc];
  int mi = blockIdx.x / Bc;
  int b  = blockIdx.x % Bc;
  const float* hb = h + b*Nc*Dc;

  if (threadIdx.x < 32){                       // wave 0 drives the TDM
    tdm_load_f32(hb, sh, Nc*Dc);
    __builtin_amdgcn_s_wait_tensorcnt(0);
  }

  int wid = threadIdx.x >> 5, lane = threadIdx.x & 31;
  int d0 = lane, d1 = lane + 32;
  int q1 = lane + 32;
  bool valid1 = (q1 < Nc);
  float sj0 = sJ[b*Nc + lane];
  float sj1 = valid1 ? sJ[b*Nc + q1] : 0.f;
  // hoist per-lane channel parameters (d0/d1 fixed per lane)
  float gb0 = gbias[d0], gb1 = gbias[d1];
  float i10 = 1.f/sqrtf(rv1[d0]+EPSc), i11 = 1.f/sqrtf(rv1[d1]+EPSc);
  float i20 = 1.f/sqrtf(rv2[d0]+EPSc), i21 = 1.f/sqrtf(rv2[d1]+EPSc);
  float rm10=rm1[d0], rm11=rm1[d1], g10=g1[d0], g11=g1[d1], b10=b1[d0], b11=b1[d1];
  float rm20=rm2[d0], rm21=rm2[d1], g20=g2[d0], g21=g2[d1], b20=b2[d0], b21=b2[d1];
  float ow0 = outw[d0], ow1 = outw[d1];
  float ob  = outb[0];
  __syncthreads();

  for (int p0 = 0; p0 < 56; p0 += 8){
    int p = p0 + wid;
    if (p < Nc){                                   // wave-uniform condition
      int s   = mi - 1;
      int qrm = (mi == 0) ? -1 : ((s < p) ? s : s + 1);
      float sip = sI[b*Nc + p];
      bool use0 = (lane != qrm);
      bool use1 = valid1 && (q1 != qrm);
      float l0 = lrelu(sip + sj0), l1 = lrelu(sip + sj1);
      float m  = wmaxr(fmaxf(use0 ? l0 : NEGc, use1 ? l1 : NEGc));
      float e0 = use0 ? expf(l0 - m) : 0.f;
      float e1 = use1 ? expf(l1 - m) : 0.f;
      float inv = 1.f / wsum(e0 + e1);
      float a0 = e0*inv, a1 = e1*inv;
      float acc0 = 0.f, acc1 = 0.f;
#pragma unroll
      for (int q = 0; q < Nc; ++q){
        float aq = (q < 32) ? __shfl(a0, q, 32) : __shfl(a1, q - 32, 32);
        acc0 += aq * sh[q*Dc + d0];
        acc1 += aq * sh[q*Dc + d1];
      }
      float v0 = fmaxf((acc0 + gb0 - rm10)*i10*g10 + b10, 0.f) * emb[p*Dc + d0];
      float v1 = fmaxf((acc1 + gb1 - rm11)*i11*g11 + b11, 0.f) * emb[p*Dc + d1];
      v0 = fmaxf((v0 - rm20)*i20*g20 + b20, 0.f);
      v1 = fmaxf((v1 - rm21)*i21*g21 + b21, 0.f);
      float c = wsum(v0*ow0 + v1*ow1);
      if (lane == 0) preds[mi*BNc + b*Nc + p] = c + ob;
    }
  }
}

// --------------------------------------------------------------------------
// Two-sample F test per (s,p); prune <=> pval > 0.05 <=> fstat < FCRIT.
// One wave per (s,p), reduce over B=128 (4 values/lane).
// --------------------------------------------------------------------------
__global__ void k_ftest(const float* __restrict__ labels,
                        const float* __restrict__ preds,
                        float* __restrict__ prune){
  int wid = threadIdx.x >> 5, lane = threadIdx.x & 31;
  int wg = blockIdx.x*8 + wid;
  if (wg >= Sc*Nc) return;
  int s = wg / Nc, p = wg % Nc;
  float sa = 0.f, qa = 0.f, sb = 0.f, qb = 0.f;
#pragma unroll
  for (int j = 0; j < 4; ++j){
    int bb  = lane + 32*j;
    int idx = bb*Nc + p;
    float lab = labels[idx];
    float av = lab - preds[(s + 1)*BNc + idx];
    float bv = lab - preds[idx];
    sa += av; qa += av*av; sb += bv; qb += bv*bv;
  }
  sa = wsum(sa); qa = wsum(qa); sb = wsum(sb); qb = wsum(qb);
  if (lane == 0){
    float ma = sa * (1.f/128.f), mb = sb * (1.f/128.f);
    float ssw = (qa - 128.f*ma*ma) + (qb - 128.f*mb*mb);
    float d = ma - mb;
    float ssb = 64.f * d * d;                 // B*((ma-g)^2+(mb-g)^2), g=(ma+mb)/2
    float fstat = ssb / (ssw * (1.f/254.f));
    prune[wg] = (fstat < FCRIT) ? 1.f : 0.f;
  }
}

// --------------------------------------------------------------------------
// Final aggregation with pruned-edge keep mask (pre-BN output stored raw).
// keep(q->p) = (q==p) || prune[s(q,p)][p]==0, s(q,p) = q<p ? q : q-1.
// One block per batch b (TDM-staged h_b); 8 waves cover p.
// --------------------------------------------------------------------------
__global__ void k_aggfinal(const float* __restrict__ h,  const float* __restrict__ sI,
                           const float* __restrict__ sJ, const float* __restrict__ prune,
                           const float* __restrict__ gbias, float* __restrict__ gout){
  __shared__ float sh[Nc*Dc];
  int b = blockIdx.x;
  const float* hb = h + b*Nc*Dc;

  if (threadIdx.x < 32){
    tdm_load_f32(hb, sh, Nc*Dc);
    __builtin_amdgcn_s_wait_tensorcnt(0);
  }

  int wid = threadIdx.x >> 5, lane = threadIdx.x & 31;
  int d0 = lane, d1 = lane + 32;
  int q1 = lane + 32;
  bool valid1 = (q1 < Nc);
  float sj0 = sJ[b*Nc + lane];
  float sj1 = valid1 ? sJ[b*Nc + q1] : 0.f;
  float gb0 = gbias[d0], gb1 = gbias[d1];
  __syncthreads();

  for (int p0 = 0; p0 < 56; p0 += 8){
    int p = p0 + wid;
    if (p < Nc){
      float sip = sI[b*Nc + p];
      bool keep0 = (lane == p) || (prune[((lane < p) ? lane : lane - 1)*Nc + p] == 0.f);
      bool keep1 = valid1 && ((q1 == p) || (prune[((q1 < p) ? q1 : q1 - 1)*Nc + p] == 0.f));
      float l0 = lrelu(sip + sj0), l1 = lrelu(sip + sj1);
      float m  = wmaxr(fmaxf(keep0 ? l0 : NEGc, keep1 ? l1 : NEGc));
      float e0 = keep0 ? expf(l0 - m) : 0.f;
      float e1 = keep1 ? expf(l1 - m) : 0.f;
      float inv = 1.f / wsum(e0 + e1);
      float a0 = e0*inv, a1 = e1*inv;
      float acc0 = 0.f, acc1 = 0.f;
#pragma unroll
      for (int q = 0; q < Nc; ++q){
        float aq = (q < 32) ? __shfl(a0, q, 32) : __shfl(a1, q - 32, 32);
        acc0 += aq * sh[q*Dc + d0];
        acc1 += aq * sh[q*Dc + d1];
      }
      int row = b*Nc + p;
      gout[row*Dc + d0] = acc0 + gb0;
      gout[row*Dc + d1] = acc1 + gb1;
    }
  }
}

// --------------------------------------------------------------------------
// Per-column mean/var over 6400 rows (train-mode batchnorm stats).
// --------------------------------------------------------------------------
__global__ void k_colstats(const float* __restrict__ X,
                           float* __restrict__ mu, float* __restrict__ var){
  int d = blockIdx.x;
  float s = 0.f, ss = 0.f;
  for (int r = threadIdx.x; r < BNc; r += 256){
    float v = X[r*Dc + d];
    s += v; ss += v*v;
  }
  __shared__ float bs[256], bq[256];
  bs[threadIdx.x] = s; bq[threadIdx.x] = ss;
  __syncthreads();
  for (int o = 128; o; o >>= 1){
    if (threadIdx.x < o){ bs[threadIdx.x] += bs[threadIdx.x+o]; bq[threadIdx.x] += bq[threadIdx.x+o]; }
    __syncthreads();
  }
  if (threadIdx.x == 0){
    float m = bs[0] * (1.f/BNc);
    mu[d]  = m;
    var[d] = bq[0] * (1.f/BNc) - m*m;
  }
}

// --------------------------------------------------------------------------
// x3 = relu(BN1_train(gout)) * emb[p]
// --------------------------------------------------------------------------
__global__ void k_x3(const float* __restrict__ gout, const float* __restrict__ mu1,
                     const float* __restrict__ vr1,  const float* __restrict__ g1,
                     const float* __restrict__ b1,   const float* __restrict__ emb,
                     float* __restrict__ x3){
  int idx = blockIdx.x*256 + threadIdx.x;
  if (idx >= BNc*Dc) return;
  int row = idx >> 6, d = idx & 63;
  int p = row % Nc;
  float v = (gout[idx] - mu1[d]) * (1.f/sqrtf(vr1[d]+EPSc)) * g1[d] + b1[d];
  x3[idx] = fmaxf(v, 0.f) * emb[p*Dc + d];
}

// --------------------------------------------------------------------------
// out[row] = relu(BN2_train(x3[row])) @ out_w + out_b   (one wave per row)
// --------------------------------------------------------------------------
__global__ void k_out(const float* __restrict__ x3, const float* __restrict__ mu2,
                      const float* __restrict__ vr2, const float* __restrict__ g2,
                      const float* __restrict__ b2,  const float* __restrict__ outw,
                      const float* __restrict__ outb, float* __restrict__ out){
  int wid = threadIdx.x >> 5, lane = threadIdx.x & 31;
  int row = blockIdx.x*8 + wid;
  int d0 = lane, d1 = lane + 32;
  float v0 = (x3[row*Dc + d0] - mu2[d0]) * (1.f/sqrtf(vr2[d0]+EPSc)) * g2[d0] + b2[d0];
  float v1 = (x3[row*Dc + d1] - mu2[d1]) * (1.f/sqrtf(vr2[d1]+EPSc)) * g2[d1] + b2[d1];
  float c = wsum(fmaxf(v0, 0.f)*outw[d0] + fmaxf(v1, 0.f)*outw[d1]);
  if (lane == 0) out[row] = c + outb[0];
}

// --------------------------------------------------------------------------
extern "C" void kernel_launch(void* const* d_in, const int* in_sizes, int n_in,
                              void* d_out, int out_size, void* d_ws, size_t ws_size,
                              hipStream_t stream){
  const float* data   = (const float*)d_in[0];
  const float* labels = (const float*)d_in[1];
  // d_in[2..6] (edge_index, src_b, dst_b, submodel_masks, removed_edge_id)
  // are reconstructed analytically and never read.
  const float* emb   = (const float*)d_in[7];
  const float* Wlin  = (const float*)d_in[8];
  const float* att_i = (const float*)d_in[9];
  const float* att_j = (const float*)d_in[10];
  const float* aem_i = (const float*)d_in[11];
  const float* aem_j = (const float*)d_in[12];
  const float* gbias = (const float*)d_in[13];
  const float* g1  = (const float*)d_in[14]; const float* b1  = (const float*)d_in[15];
  const float* rm1 = (const float*)d_in[16]; const float* rv1 = (const float*)d_in[17];
  const float* g2  = (const float*)d_in[18]; const float* b2  = (const float*)d_in[19];
  const float* rm2 = (const float*)d_in[20]; const float* rv2 = (const float*)d_in[21];
  const float* outw = (const float*)d_in[22];
  const float* outb = (const float*)d_in[23];

  float* ws   = (float*)d_ws;
  float* h    = ws;               // 409600
  float* sI   = ws +  409600;     //   6400
  float* sJ   = ws +  416000;     //   6400
  float* eI   = ws +  422400;     //     64
  float* eJ   = ws +  422464;     //     64
  float* preds= ws +  422528;     // 320000
  float* prune= ws +  742528;     //   2450
  float* gout = ws +  745024;     // 409600
  float* x3   = ws + 1154624;     // 409600
  float* mu1  = ws + 1564224;     //     64
  float* vr1  = ws + 1564288;     //     64
  float* mu2  = ws + 1564352;     //     64
  float* vr2  = ws + 1564416;     //     64   (total ~6.3 MB)

  k_embdot <<<1,    64, 0, stream>>>(emb, aem_i, aem_j, eI, eJ);
  k_gemm_h <<<100, 128, 0, stream>>>(data, Wlin, h);
  k_sij    <<<25,  256, 0, stream>>>(h, att_i, att_j, eI, eJ, sI, sJ);
  k_preds  <<<NMASK*Bc, 256, 0, stream>>>(h, sI, sJ, gbias,
                                          g1, b1, rm1, rv1, g2, b2, rm2, rv2,
                                          emb, outw, outb, preds);
  k_ftest  <<<(Sc*Nc + 7)/8, 256, 0, stream>>>(labels, preds, prune);
  k_aggfinal<<<Bc, 256, 0, stream>>>(h, sI, sJ, prune, gbias, gout);
  k_colstats<<<Dc, 256, 0, stream>>>(gout, mu1, vr1);
  k_x3     <<<(BNc*Dc)/256, 256, 0, stream>>>(gout, mu1, vr1, g1, b1, emb, x3);
  k_colstats<<<Dc, 256, 0, stream>>>(x3, mu2, vr2);
  k_out    <<<BNc/8, 256, 0, stream>>>(x3, mu2, vr2, g2, b2, outw, outb, (float*)d_out);
}
